// Encoder_5978594476039
// MI455X (gfx1250) — matrix operand order repaired and checked by reading, not compiled, not run on hardware
//
#include <hip/hip_runtime.h>

// ---------------------------------------------------------------------------
// 2-layer LSTM on MI455X (gfx1250): persistent-kernel bf16 WMMA design.
//
// Roofline: 0.89 TFLOP total across 512 *serial* steps. Weights are 27 MB in
// bf16 -> L2-resident (192 MB L2); per-step latency, not HBM bandwidth, is the
// limit. So: one persistent kernel (32 blocks, trivially co-resident), grid
// barriers instead of 1024 kernel launches, LSTM cell state c1/c2 held in
// VGPRs (WMMA D-layout) for the whole sequence, bias preloaded to registers.
// Each wave owns a 16-batch x 16-unit tile across all 4 gates.
// ---------------------------------------------------------------------------

typedef __attribute__((ext_vector_type(16))) __bf16 v16bf;
typedef __attribute__((ext_vector_type(8)))  __bf16 v8bf;
typedef __attribute__((ext_vector_type(8)))  float  v8f;

#define UNITS   1024
#define GATES   4096   // 4*UNITS
#define BATCH   64
#define SEQ     512
#define EMBD    256
#define KCHUNK  128    // LDS-staged K span for the A (activation) tile
#define NBLOCKS 32     // grid(8,4) -- must match launch below

static __device__ __forceinline__ v16bf make_frag(const __bf16* p0,
                                                  const __bf16* p1) {
  v8bf lo = *(const v8bf*)p0;
  v8bf hi = *(const v8bf*)p1;
  return __builtin_shufflevector(lo, hi, 0, 1, 2, 3, 4, 5, 6, 7,
                                         8, 9, 10, 11, 12, 13, 14, 15);
}

// acc[g] += A[m0:m0+16, :K] * W[g*1024+u0 : +16, :K]^T ; A staged via LDS.
static __device__ __forceinline__ void gemm_acc(
    const __bf16* __restrict__ A, int K, const __bf16* __restrict__ Wt,
    int m0, int u0, int lrow, int lhi, __bf16* smem, v8f* acc) {
  const int tid   = threadIdx.x;
  const int koffA = lhi * 8;    // A frag: lanes 16-31 take K octets +8 / +24
  const int koffB = lhi * 8;    // B frag: lanes 16-31 take K octets +8 / +24
  for (int kk0 = 0; kk0 < K; kk0 += KCHUNK) {
    __syncthreads();            // protect previous chunk's LDS reads
    {
      const int lin = tid * 8;  // 256 threads x 8 bf16 = 16 x 128 tile
      const int r   = lin >> 7;
      const int c   = lin & (KCHUNK - 1);
      *(v8bf*)(smem + r * KCHUNK + c) =
          *(const v8bf*)(A + (size_t)(m0 + r) * K + kk0 + c);
    }
    __syncthreads();
#pragma unroll
    for (int kc = 0; kc < KCHUNK; kc += 32) {
      const __bf16* ap = smem + lrow * KCHUNK + kc + koffA;
      v16bf afrag = make_frag(ap, ap + 16);
#pragma unroll
      for (int g = 0; g < 4; ++g) {
        const __bf16* wp =
            Wt + (size_t)(g * UNITS + u0 + lrow) * K + kk0 + kc + koffB;
        v16bf bfrag = make_frag(wp, wp + 16);
        acc[g] = __builtin_amdgcn_wmma_f32_16x16x32_bf16(
            false, afrag, false, bfrag, (short)0, acc[g], false, false);
      }
    }
  }
}

// Device-wide barrier: release fence -> arrive -> acquire poll -> fence.
static __device__ __forceinline__ void grid_barrier(unsigned* bar,
                                                    unsigned gen) {
  __threadfence();              // flush this thread's h stores to L2 (device)
  __syncthreads();
  if (threadIdx.x == 0) {
    __hip_atomic_fetch_add(bar, 1u, __ATOMIC_RELEASE,
                           __HIP_MEMORY_SCOPE_AGENT);
    while (__hip_atomic_load(bar, __ATOMIC_ACQUIRE,
                             __HIP_MEMORY_SCOPE_AGENT) < gen * NBLOCKS) {
      __builtin_amdgcn_s_sleep(1);
    }
  }
  __syncthreads();
  __threadfence();              // invalidate WGP cache: see peers' h writes
}

// One LSTM layer for one time step; c kept in caller's registers.
static __device__ __forceinline__ void layer_tick(
    const __bf16* __restrict__ A0, int K0, const __bf16* __restrict__ W0,
    const __bf16* __restrict__ A1, int K1, const __bf16* __restrict__ W1,
    float bi, float bf_, float bg, float bo, v8f& creg,
    __bf16* __restrict__ h_out, float* __restrict__ f32_out,
    long long out_stride, int m0, int u0, int lrow, int lhi, __bf16* smem) {
  v8f acc[4] = {};   // i, f, g, o
  gemm_acc(A0, K0, W0, m0, u0, lrow, lhi, smem, acc);  // x @ Wk
  gemm_acc(A1, K1, W1, m0, u0, lrow, lhi, smem, acc);  // h @ Wr
  const int u = u0 + lrow;
#pragma unroll
  for (int r = 0; r < 8; ++r) {
    const int m = m0 + r + lhi * 8;     // D layout: lanes>=16 -> M+8
    const float zi = acc[0][r] + bi;
    const float zf = acc[1][r] + bf_;
    const float zg = acc[2][r] + bg;
    const float zo = acc[3][r] + bo;
    const float si = 1.f / (1.f + __expf(-zi));
    const float sf = 1.f / (1.f + __expf(-zf));
    const float so = 1.f / (1.f + __expf(-zo));
    const float cn = sf * creg[r] + si * tanhf(zg);
    const float hn = so * tanhf(cn);
    creg[r] = cn;
    h_out[m * UNITS + u] = (__bf16)hn;
    if (f32_out) f32_out[(long long)m * out_stride + u] = hn;
  }
}

// Persistent kernel: all 512 steps, both layers, grid-synced via atomics.
// grid = (8,4): blockIdx.x -> 128-unit span, blockIdx.y -> 16-row batch tile.
__global__ __launch_bounds__(256)
void lstm_persistent_kernel(const __bf16* __restrict__ xe,
                            const __bf16* __restrict__ WkT1,
                            const __bf16* __restrict__ WrT1,
                            const float* __restrict__ b1,
                            const __bf16* __restrict__ WkT2,
                            const __bf16* __restrict__ WrT2,
                            const float* __restrict__ b2,
                            __bf16* __restrict__ h1,
                            __bf16* __restrict__ h2,
                            float* __restrict__ out,
                            unsigned* __restrict__ bar) {
  __shared__ __attribute__((aligned(16))) __bf16 smem[16 * KCHUNK];
  const int tid  = threadIdx.x;
  const int wave = tid >> 5;
  const int lane = tid & 31;
  const int lrow = lane & 15;
  const int lhi  = lane >> 4;
  const int m0   = blockIdx.y * 16;
  const int u0   = blockIdx.x * 128 + wave * 16;
  const int u    = u0 + lrow;
  const int HS   = BATCH * UNITS;

  // Bias in registers for the whole sequence.
  const float b1i = b1[u], b1f = b1[UNITS + u];
  const float b1g = b1[2 * UNITS + u], b1o = b1[3 * UNITS + u];
  const float b2i = b2[u], b2f = b2[UNITS + u];
  const float b2g = b2[2 * UNITS + u], b2o = b2[3 * UNITS + u];

  v8f c1r = {};      // cell states live in VGPRs (WMMA D layout) all sequence
  v8f c2r = {};
  unsigned gen = 0;

  for (int t = 0; t < SEQ; ++t) {
    const __bf16* h1_in  = h1 + (t & 1) * HS;
    __bf16*       h1_out = h1 + ((t + 1) & 1) * HS;
    const __bf16* h2_in  = h2 + (t & 1) * HS;
    __bf16*       h2_out = h2 + ((t + 1) & 1) * HS;

    layer_tick(xe + (size_t)t * BATCH * EMBD, EMBD, WkT1,
               h1_in, UNITS, WrT1, b1i, b1f, b1g, b1o, c1r,
               h1_out, nullptr, 0, m0, u0, lrow, lhi, smem);
    grid_barrier(bar, ++gen);               // h1(t) visible everywhere

    layer_tick(h1_out, UNITS, WkT2,
               h2_in, UNITS, WrT2, b2i, b2f, b2g, b2o, c2r,
               h2_out, out + (size_t)t * UNITS, (long long)SEQ * UNITS,
               m0, u0, lrow, lhi, smem);
    grid_barrier(bar, ++gen);               // h2(t) visible everywhere
  }
}

// W [K][4096] fp32 -> Wt [4096][K] bf16
__global__ void cvt_transpose_kernel(const float* __restrict__ W,
                                     __bf16* __restrict__ Wt, int K,
                                     long long total) {
  const long long idx = (long long)blockIdx.x * blockDim.x + threadIdx.x;
  if (idx >= total) return;
  const int n = (int)(idx % GATES);
  const int k = (int)(idx / GATES);
  Wt[(size_t)n * K + k] = (__bf16)W[idx];
}

// xe[t][b][e] = bf16(emb[x[b][t]][e])
__global__ void embed_kernel(const int* __restrict__ x,
                             const float* __restrict__ emb,
                             __bf16* __restrict__ xe) {
  const long long idx = (long long)blockIdx.x * blockDim.x + threadIdx.x;
  if (idx >= (long long)SEQ * BATCH * EMBD) return;
  const int e = (int)(idx % EMBD);
  const int b = (int)((idx / EMBD) % BATCH);
  const int t = (int)(idx / ((long long)EMBD * BATCH));
  const int tok = x[b * SEQ + t];
  xe[idx] = (__bf16)emb[(size_t)tok * EMBD + e];
}

// Zero both ping-pong h buffers (2n each) and the grid-barrier counter.
__global__ void zero_state_kernel(__bf16* h1, __bf16* h2, unsigned* bar,
                                  int n) {
  const int idx = blockIdx.x * blockDim.x + threadIdx.x;
  if (idx < 2 * n) { h1[idx] = (__bf16)0.f; h2[idx] = (__bf16)0.f; }
  if (idx == 0) *bar = 0u;
}

extern "C" void kernel_launch(void* const* d_in, const int* in_sizes, int n_in,
                              void* d_out, int out_size, void* d_ws,
                              size_t ws_size, hipStream_t stream) {
  (void)in_sizes; (void)n_in; (void)out_size; (void)ws_size;
  const int*   x   = (const int*)d_in[0];
  const float* emb = (const float*)d_in[1];
  const float* Wk1 = (const float*)d_in[2];
  const float* Wr1 = (const float*)d_in[3];
  const float* b1  = (const float*)d_in[4];
  const float* Wk2 = (const float*)d_in[5];
  const float* Wr2 = (const float*)d_in[6];
  const float* b2  = (const float*)d_in[7];
  float* out = (float*)d_out;

  char*  ws  = (char*)d_ws;
  size_t off = 0;
  auto carve = [&](size_t bytes) -> void* {
    void* p = ws + off;
    off = (off + bytes + 255) & ~(size_t)255;
    return p;
  };
  __bf16* WkT1 = (__bf16*)carve((size_t)GATES * EMBD * 2);       //  2 MB
  __bf16* WrT1 = (__bf16*)carve((size_t)GATES * UNITS * 2);      //  8 MB
  __bf16* WkT2 = (__bf16*)carve((size_t)GATES * UNITS * 2);      //  8 MB
  __bf16* WrT2 = (__bf16*)carve((size_t)GATES * UNITS * 2);      //  8 MB
  __bf16* xe   = (__bf16*)carve((size_t)SEQ * BATCH * EMBD * 2); // 16.8 MB
  __bf16* h1   = (__bf16*)carve((size_t)2 * BATCH * UNITS * 2);  // ping-pong
  __bf16* h2   = (__bf16*)carve((size_t)2 * BATCH * UNITS * 2);
  unsigned* bar = (unsigned*)carve(256);

  const int HS = BATCH * UNITS;

  // One-time prep (deterministic, every launch).
  const long long tk1 = (long long)EMBD * GATES;
  const long long tr  = (long long)UNITS * GATES;
  cvt_transpose_kernel<<<(unsigned)((tk1 + 255) / 256), 256, 0, stream>>>(
      Wk1, WkT1, EMBD, tk1);
  cvt_transpose_kernel<<<(unsigned)((tr + 255) / 256), 256, 0, stream>>>(
      Wr1, WrT1, UNITS, tr);
  cvt_transpose_kernel<<<(unsigned)((tr + 255) / 256), 256, 0, stream>>>(
      Wk2, WkT2, UNITS, tr);
  cvt_transpose_kernel<<<(unsigned)((tr + 255) / 256), 256, 0, stream>>>(
      Wr2, WrT2, UNITS, tr);
  const long long te = (long long)SEQ * BATCH * EMBD;
  embed_kernel<<<(unsigned)((te + 255) / 256), 256, 0, stream>>>(x, emb, xe);
  zero_state_kernel<<<(2 * HS + 255) / 256, 256, 0, stream>>>(h1, h2, bar, HS);

  // Single persistent kernel runs the whole sequence.
  lstm_persistent_kernel<<<dim3(8, 4), 256, 0, stream>>>(
      xe, WkT1, WrT1, b1, WkT2, WrT2, b2, h1, h2, out, bar);
}
